// UltimateRAv3_44100724195817
// MI455X (gfx1250) — compile-verified, matrix operand order checked by reading, not run on hardware
//
#include <hip/hip_runtime.h>

typedef __bf16 bf16;
typedef __attribute__((ext_vector_type(16))) __bf16 v16bf;
typedef __attribute__((ext_vector_type(8)))  __bf16 v8bf;
typedef __attribute__((ext_vector_type(8)))  float  v8f;
typedef __attribute__((ext_vector_type(4)))  int    v4i;

// Problem constants (from reference)
#define B_DIM 2
#define T_DIM 2048
#define C_DIM 2048
#define H_DIM 16
#define HD    128      // head_dim
#define R_DIM 8
#define DSTD  120      // head_dim - R
#define F_DIM 6144     // 3*C
#define M_DIM 4096     // B*T

// CDNA5 async global->LDS path (ASYNCcnt-tracked), if the toolchain exposes it.
#if defined(__HIP_DEVICE_COMPILE__) && defined(__has_builtin)
#if __has_builtin(__builtin_amdgcn_global_load_async_to_lds_b128)
#define HAVE_ASYNC_LDS 1
#endif
#endif
#ifndef HAVE_ASYNC_LDS
#define HAVE_ASYNC_LDS 0
#endif

#if HAVE_ASYNC_LDS
__device__ __forceinline__ void async_copy_b128(const bf16* g, bf16* l) {
  // Signature (from hipcc diagnostic): param 0/1 are pointers to v4i (16B payload).
  __builtin_amdgcn_global_load_async_to_lds_b128((v4i*)g, (v4i*)l, 0, 0);
}
#define WAIT_ASYNC() asm volatile("s_wait_asynccnt 0" ::: "memory")
#else
#define WAIT_ASYNC()
#endif

// ---------------------------------------------------------------------------
// f32 -> bf16 convert (grid-stride)
// ---------------------------------------------------------------------------
__global__ void cvt_f32_bf16(const float* __restrict__ src, bf16* __restrict__ dst,
                             size_t n) {
  size_t i = (size_t)blockIdx.x * blockDim.x + threadIdx.x;
  size_t stride = (size_t)gridDim.x * blockDim.x;
  for (; i < n; i += stride) dst[i] = (bf16)src[i];
}

// ---------------------------------------------------------------------------
// WMMA GEMM: out[m][n] = sum_k A[m][k] * W[n][k]   (A @ W^T)
// A: [Mdim x Kdim] bf16 row-major, W: [Ndim x Kdim] bf16 row-major.
// Block = 256 threads = 8 waves; tile 128(M) x 128(N); K-step 32.
// Double-buffered LDS; async global->LDS when available, else reg staging.
// ---------------------------------------------------------------------------
#define GSTR 40  // LDS row stride in halves (32 data + 8 pad) -> 80B rows, conflict-free

template <typename OutT>
__device__ inline OutT cvt_out(float v);
template <> __device__ inline float cvt_out<float>(float v) { return v; }
template <> __device__ inline bf16  cvt_out<bf16>(float v)  { return (bf16)v; }

template <typename OutT>
__global__ __launch_bounds__(256)
void gemm_wmma(const bf16* __restrict__ A, const bf16* __restrict__ W,
               OutT* __restrict__ out, int Mdim, int Ndim, int Kdim) {
  __shared__ bf16 As[2][128 * GSTR];
  __shared__ bf16 Bs[2][128 * GSTR];

  const int tid  = threadIdx.x;
  const int lane = tid & 31;
  const int wave = tid >> 5;
  const int wm   = wave >> 2;   // 0..1
  const int wn   = wave & 3;    // 0..3
  const int g    = lane >> 4;   // half-wave
  const int lr   = lane & 15;
  const int mBase = blockIdx.y * 128;
  const int nBase = blockIdx.x * 128;

  // Per-thread cooperative-load coordinates: 2 chunks per matrix per tile.
  const int lrow = tid >> 2;          // 0..63 (+64 for c=1)
  const int lcol = (tid & 3) * 8;     // 0,8,16,24

  v8f acc[4][2];
#pragma unroll
  for (int mt = 0; mt < 4; ++mt)
#pragma unroll
    for (int nt = 0; nt < 2; ++nt)
#pragma unroll
      for (int v = 0; v < 8; ++v) acc[mt][nt][v] = 0.f;

  const int nk = Kdim / 32;
  int cur = 0;

#if HAVE_ASYNC_LDS
  // Prologue: async-load tile 0 into buffer 0.
#pragma unroll
  for (int c = 0; c < 2; ++c) {
    const int row = lrow + c * 64;
    async_copy_b128(A + (size_t)(mBase + row) * Kdim + lcol, &As[0][row * GSTR + lcol]);
    async_copy_b128(W + (size_t)(nBase + row) * Kdim + lcol, &Bs[0][row * GSTR + lcol]);
  }
#else
  v8bf pa[2], pb[2];
#pragma unroll
  for (int c = 0; c < 2; ++c) {
    const int row = lrow + c * 64;
    pa[c] = *(const v8bf*)(A + (size_t)(mBase + row) * Kdim + lcol);
    pb[c] = *(const v8bf*)(W + (size_t)(nBase + row) * Kdim + lcol);
  }
#pragma unroll
  for (int c = 0; c < 2; ++c) {
    const int row = lrow + c * 64;
    *(v8bf*)(&As[0][row * GSTR + lcol]) = pa[c];
    *(v8bf*)(&Bs[0][row * GSTR + lcol]) = pb[c];
  }
  __syncthreads();
#endif

  for (int it = 0; it < nk; ++it) {
#if HAVE_ASYNC_LDS
    WAIT_ASYNC();          // this wave's oldest (current-tile) copies done
    __syncthreads();       // -> all waves' copies done; prior reads retired
    if (it + 1 < nk) {
      const int k0n = (it + 1) * 32;
#pragma unroll
      for (int c = 0; c < 2; ++c) {
        const int row = lrow + c * 64;
        async_copy_b128(A + (size_t)(mBase + row) * Kdim + k0n + lcol,
                        &As[cur ^ 1][row * GSTR + lcol]);
        async_copy_b128(W + (size_t)(nBase + row) * Kdim + k0n + lcol,
                        &Bs[cur ^ 1][row * GSTR + lcol]);
      }
    }
#else
    if (it + 1 < nk) {
      const int k0n = (it + 1) * 32;
#pragma unroll
      for (int c = 0; c < 2; ++c) {
        const int row = lrow + c * 64;
        pa[c] = *(const v8bf*)(A + (size_t)(mBase + row) * Kdim + k0n + lcol);
        pb[c] = *(const v8bf*)(W + (size_t)(nBase + row) * Kdim + k0n + lcol);
      }
    }
#endif

    // ---- compute current buffer ----
    v16bf afr[4], bfr[2];
#pragma unroll
    for (int mt = 0; mt < 4; ++mt) {
      const bf16* p = &As[cur][(wm * 64 + mt * 16 + lr) * GSTR + g * 8];
      v8bf lo = *(const v8bf*)(p);
      v8bf hi = *(const v8bf*)(p + 16);
#pragma unroll
      for (int i = 0; i < 8; ++i) { afr[mt][i] = lo[i]; afr[mt][i + 8] = hi[i]; }
    }
#pragma unroll
    for (int nt = 0; nt < 2; ++nt) {
      const bf16* p = &Bs[cur][(wn * 32 + nt * 16 + lr) * GSTR + g * 8];
      v8bf lo = *(const v8bf*)(p);
      v8bf hi = *(const v8bf*)(p + 16);
#pragma unroll
      for (int i = 0; i < 8; ++i) { bfr[nt][i] = lo[i]; bfr[nt][i + 8] = hi[i]; }
    }
#pragma unroll
    for (int mt = 0; mt < 4; ++mt)
#pragma unroll
      for (int nt = 0; nt < 2; ++nt)
        acc[mt][nt] = __builtin_amdgcn_wmma_f32_16x16x32_bf16(
            false, afr[mt], false, bfr[nt], (short)0, acc[mt][nt], false, false);

#if !HAVE_ASYNC_LDS
    if (it + 1 < nk) {
#pragma unroll
      for (int c = 0; c < 2; ++c) {
        const int row = lrow + c * 64;
        *(v8bf*)(&As[cur ^ 1][row * GSTR + lcol]) = pa[c];
        *(v8bf*)(&Bs[cur ^ 1][row * GSTR + lcol]) = pb[c];
      }
    }
    __syncthreads();
#endif
    cur ^= 1;
  }

  // Epilogue: D layout row = g*8+v, col = lr
#pragma unroll
  for (int mt = 0; mt < 4; ++mt)
#pragma unroll
    for (int nt = 0; nt < 2; ++nt)
#pragma unroll
      for (int v = 0; v < 8; ++v) {
        const int row = mBase + wm * 64 + mt * 16 + g * 8 + v;
        const int col = nBase + wn * 32 + nt * 16 + lr;
        out[(size_t)row * Ndim + col] = cvt_out<OutT>(acc[mt][nt][v]);
      }
}

// ---------------------------------------------------------------------------
// Repack fused output into Q/K/V [B,H,T,128] bf16 with per-head routing.
// ---------------------------------------------------------------------------
__global__ void repack_kernel(const bf16* __restrict__ fused,
                              const float* __restrict__ w_std,
                              const float* __restrict__ w_rec,
                              bf16* __restrict__ Qf, bf16* __restrict__ Kf,
                              bf16* __restrict__ Vv) {
  const size_t idx = (size_t)blockIdx.x * blockDim.x + threadIdx.x;
  const size_t total = (size_t)M_DIM * F_DIM;
  if (idx >= total) return;
  const int    f = (int)(idx % F_DIM);
  const size_t m = idx / F_DIM;
  const int b = (int)(m / T_DIM);
  const int t = (int)(m % T_DIM);
  const float val = (float)fused[idx];

  int h, d;
  bf16* dst;
  float scl;
  if (f < 1920) {                       // Q_std
    h = f / DSTD; d = f % DSTD;
    scl = sqrtf(fmaxf(w_std[h], 1e-8f)); dst = Qf;
  } else if (f < 3840) {                // K_std
    const int f2 = f - 1920; h = f2 / DSTD; d = f2 % DSTD;
    scl = sqrtf(fmaxf(w_std[h], 1e-8f)); dst = Kf;
  } else if (f < 5888) {                // V
    const int f2 = f - 3840; h = f2 / HD; d = f2 % HD;
    scl = 1.f; dst = Vv;
  } else if (f < 6016) {                // Q_low
    const int f2 = f - 5888; h = f2 / R_DIM; d = DSTD + (f2 % R_DIM);
    const bool ra = w_rec[h] > 0.1f;
    scl = ra ? sqrtf(fmaxf(w_rec[h], 1e-8f)) : sqrtf(fmaxf(w_std[h], 1e-8f));
    dst = ra ? Kf : Qf;                 // RA: Q_low feeds Kf_low
  } else {                              // K_low
    const int f2 = f - 6016; h = f2 / R_DIM; d = DSTD + (f2 % R_DIM);
    const bool ra = w_rec[h] > 0.1f;
    scl = ra ? sqrtf(fmaxf(w_rec[h], 1e-8f)) : sqrtf(fmaxf(w_std[h], 1e-8f));
    dst = ra ? Qf : Kf;                 // RA: K_low feeds Qf_low
  }
  dst[(((size_t)b * H_DIM + h) * T_DIM + t) * HD + d] = (bf16)(val * scl);
}

// ---------------------------------------------------------------------------
// Flash attention (causal), bf16 WMMA, f32 accumulate, online softmax.
// Grid: (T/64, H, B). Block: 128 threads = 4 waves; wave w owns 16 q-rows.
// Out: Ob[b*T + t][h*128 + d] bf16 (ready for the projection GEMM).
// ---------------------------------------------------------------------------
#define KSTR 136  // 128 + 8 pad halves -> 272B rows
#define VSTR 72   // 64 + 8 pad halves  -> 144B rows
#define PSTR 72

__global__ __launch_bounds__(128)
void flash_attn(const bf16* __restrict__ Qf, const bf16* __restrict__ Kf,
                const bf16* __restrict__ Vv, bf16* __restrict__ Ob) {
  __shared__ bf16 Ksmem[64 * KSTR];      // K tile row-major [key][d]
  __shared__ bf16 Vsmem[128 * VSTR];     // V^T tile       [d][key]
  __shared__ bf16 Psmem[4 * 16 * PSTR];  // per-wave P relayout

  const int tid  = threadIdx.x;
  const int lane = tid & 31;
  const int w    = tid >> 5;   // wave 0..3
  const int g    = lane >> 4;
  const int lr   = lane & 15;
  const int qb   = blockIdx.x;
  const int h    = blockIdx.y;
  const int b    = blockIdx.z;

  const float scale = 0.08838834764831845f;  // 1/sqrt(128)
  const size_t headBase = ((size_t)b * H_DIM + h) * T_DIM;

  // Q fragments (16 rows x 128 d = 4 k-frags), loaded once from global.
  v16bf qfrag[4];
  {
    const bf16* Qp = Qf + (headBase + (size_t)qb * 64 + w * 16 + lr) * HD;
#pragma unroll
    for (int kf = 0; kf < 4; ++kf) {
      const bf16* qp = Qp + kf * 32 + g * 8;
      v8bf lo = *(const v8bf*)qp;
      v8bf hi = *(const v8bf*)(qp + 16);
#pragma unroll
      for (int i = 0; i < 8; ++i) { qfrag[kf][i] = lo[i]; qfrag[kf][i + 8] = hi[i]; }
    }
  }

  v8f o[8];
#pragma unroll
  for (int ot = 0; ot < 8; ++ot)
#pragma unroll
    for (int v = 0; v < 8; ++v) o[ot][v] = 0.f;
  float rmax[8], rsum[8];
#pragma unroll
  for (int v = 0; v < 8; ++v) { rmax[v] = -__builtin_inff(); rsum[v] = 0.f; }

  const int qRowBase = qb * 64 + w * 16 + g * 8;

  for (int kb = 0; kb <= qb; ++kb) {
    const int kBase = kb * 64;
    __syncthreads();
    const bf16* Kp = Kf + (headBase + kBase) * HD;
    const bf16* Vp = Vv + (headBase + kBase) * HD;
    // K tile: async copy straight into LDS when available.
#pragma unroll
    for (int it = 0; it < 8; ++it) {
      const int chunk = tid + it * 128;
      const int row = chunk >> 4;          // key 0..63
      const int col = (chunk & 15) * 8;    // d
#if HAVE_ASYNC_LDS
      async_copy_b128(Kp + (size_t)row * HD + col, Ksmem + row * KSTR + col);
#else
      v8bf kv = *(const v8bf*)(Kp + (size_t)row * HD + col);
      *(v8bf*)(Ksmem + row * KSTR + col) = kv;
#endif
    }
    // V tile: transpose on the fly through VGPRs.
#pragma unroll
    for (int it = 0; it < 8; ++it) {
      const int chunk = tid + it * 128;
      const int row = chunk >> 4;
      const int col = (chunk & 15) * 8;
      v8bf vv8 = *(const v8bf*)(Vp + (size_t)row * HD + col);
#pragma unroll
      for (int i = 0; i < 8; ++i) Vsmem[(col + i) * VSTR + row] = vv8[i];
    }
    WAIT_ASYNC();
    __syncthreads();

    // S = Q K^T : 4 n-tiles of 16 keys, K-dim = 128 (4 WMMA steps each)
    v8f s[4];
#pragma unroll
    for (int nt = 0; nt < 4; ++nt) {
#pragma unroll
      for (int v = 0; v < 8; ++v) s[nt][v] = 0.f;
#pragma unroll
      for (int kf = 0; kf < 4; ++kf) {
        const bf16* p = Ksmem + (nt * 16 + lr) * KSTR + kf * 32 + g * 8;
        v8bf lo = *(const v8bf*)p;
        v8bf hi = *(const v8bf*)(p + 16);
        v16bf bfrg;
#pragma unroll
        for (int i = 0; i < 8; ++i) { bfrg[i] = lo[i]; bfrg[i + 8] = hi[i]; }
        s[nt] = __builtin_amdgcn_wmma_f32_16x16x32_bf16(
            false, qfrag[kf], false, bfrg, (short)0, s[nt], false, false);
      }
    }

    // Online softmax over this 16x64 score block.
    const bool diag = (kb == qb);
#pragma unroll
    for (int v = 0; v < 8; ++v) {
      const int qrow = qRowBase + v;
      float mx = rmax[v];
#pragma unroll
      for (int nt = 0; nt < 4; ++nt) {
        float sv = s[nt][v] * scale;
        if (diag && (kBase + nt * 16 + lr > qrow)) sv = -__builtin_inff();
        s[nt][v] = sv;
        mx = fmaxf(mx, sv);
      }
#pragma unroll
      for (int off = 1; off < 16; off <<= 1)
        mx = fmaxf(mx, __shfl_xor(mx, off, 32));
      const float alpha = __expf(rmax[v] - mx);
      rmax[v] = mx;
      float ps = 0.f;
#pragma unroll
      for (int nt = 0; nt < 4; ++nt) {
        const float p = __expf(s[nt][v] - mx);
        s[nt][v] = p;
        ps += p;
      }
#pragma unroll
      for (int off = 1; off < 16; off <<= 1)
        ps += __shfl_xor(ps, off, 32);
      rsum[v] = rsum[v] * alpha + ps;
#pragma unroll
      for (int ot = 0; ot < 8; ++ot) o[ot][v] *= alpha;
    }

    // Relayout P (C-layout) -> A-fragment layout via per-wave LDS.
    bf16* Pw = Psmem + w * (16 * PSTR);
#pragma unroll
    for (int nt = 0; nt < 4; ++nt)
#pragma unroll
      for (int v = 0; v < 8; ++v)
        Pw[(g * 8 + v) * PSTR + nt * 16 + lr] = (bf16)s[nt][v];
    asm volatile("s_wait_dscnt 0" ::: "memory");  // wave-local LDS RAW

    // O += P V : K-dim = 64 keys (2 WMMA steps), 8 output d-tiles.
#pragma unroll
    for (int pf = 0; pf < 2; ++pf) {
      const bf16* pp = Pw + lr * PSTR + pf * 32 + g * 8;
      v8bf plo = *(const v8bf*)pp;
      v8bf phi = *(const v8bf*)(pp + 16);
      v16bf afrg;
#pragma unroll
      for (int i = 0; i < 8; ++i) { afrg[i] = plo[i]; afrg[i + 8] = phi[i]; }
#pragma unroll
      for (int ot = 0; ot < 8; ++ot) {
        const bf16* vp = Vsmem + (ot * 16 + lr) * VSTR + pf * 32 + g * 8;
        v8bf vlo = *(const v8bf*)vp;
        v8bf vhi = *(const v8bf*)(vp + 16);
        v16bf bfrg;
#pragma unroll
        for (int i = 0; i < 8; ++i) { bfrg[i] = vlo[i]; bfrg[i + 8] = vhi[i]; }
        o[ot] = __builtin_amdgcn_wmma_f32_16x16x32_bf16(
            false, afrg, false, bfrg, (short)0, o[ot], false, false);
      }
    }
  }

  // Normalize and store: Ob[b*T + t][h*128 + d]
  const size_t obase = ((size_t)b * T_DIM + qb * 64 + w * 16) * C_DIM + h * HD;
#pragma unroll
  for (int ot = 0; ot < 8; ++ot)
#pragma unroll
    for (int v = 0; v < 8; ++v) {
      const float val = o[ot][v] / rsum[v];
      Ob[obase + (size_t)(g * 8 + v) * C_DIM + ot * 16 + lr] = (bf16)val;
    }
}

// ---------------------------------------------------------------------------
// Launch
// ---------------------------------------------------------------------------
extern "C" void kernel_launch(void* const* d_in, const int* in_sizes, int n_in,
                              void* d_out, int out_size, void* d_ws, size_t ws_size,
                              hipStream_t stream) {
  (void)in_sizes; (void)n_in; (void)out_size; (void)ws_size;
  const float* x      = (const float*)d_in[0];
  const float* W_attn = (const float*)d_in[1];
  const float* W_proj = (const float*)d_in[2];
  const float* w_std  = (const float*)d_in[3];
  const float* w_rec  = (const float*)d_in[4];
  float* out = (float*)d_out;

  // Workspace layout (all offsets multiple of 256B)
  char* ws = (char*)d_ws;
  size_t off = 0;
  bf16* Xb    = (bf16*)(ws + off); off += (size_t)M_DIM * C_DIM * 2;  // 16 MB
  bf16* Wa    = (bf16*)(ws + off); off += (size_t)F_DIM * C_DIM * 2;  // 24 MB
  bf16* Wp    = (bf16*)(ws + off); off += (size_t)C_DIM * C_DIM * 2;  //  8 MB
  bf16* Fused = (bf16*)(ws + off); off += (size_t)M_DIM * F_DIM * 2;  // 48 MB
  bf16* Qf    = (bf16*)(ws + off); off += (size_t)B_DIM * H_DIM * T_DIM * HD * 2;
  bf16* Kf    = (bf16*)(ws + off); off += (size_t)B_DIM * H_DIM * T_DIM * HD * 2;
  bf16* Vv    = (bf16*)(ws + off); off += (size_t)B_DIM * H_DIM * T_DIM * HD * 2;
  bf16* Ob    = (bf16*)(ws + off); off += (size_t)M_DIM * C_DIM * 2;

  // 1) f32 -> bf16 converts
  cvt_f32_bf16<<<4096, 256, 0, stream>>>(x, Xb, (size_t)M_DIM * C_DIM);
  cvt_f32_bf16<<<4096, 256, 0, stream>>>(W_attn, Wa, (size_t)F_DIM * C_DIM);
  cvt_f32_bf16<<<4096, 256, 0, stream>>>(W_proj, Wp, (size_t)C_DIM * C_DIM);

  // 2) fused = x @ W_attn^T  (4096 x 6144 x 2048)
  gemm_wmma<bf16><<<dim3(F_DIM / 128, M_DIM / 128), 256, 0, stream>>>(
      Xb, Wa, Fused, M_DIM, F_DIM, C_DIM);

  // 3) per-head scaling + RA routing into Q/K/V
  {
    const size_t total = (size_t)M_DIM * F_DIM;
    repack_kernel<<<(unsigned)((total + 255) / 256), 256, 0, stream>>>(
        Fused, w_std, w_rec, Qf, Kf, Vv);
  }

  // 4) causal flash attention
  flash_attn<<<dim3(T_DIM / 64, H_DIM, B_DIM), 128, 0, stream>>>(Qf, Kf, Vv, Ob);

  // 5) out = O @ W_proj^T  (4096 x 2048 x 2048), f32 output
  gemm_wmma<float><<<dim3(C_DIM / 128, M_DIM / 128), 256, 0, stream>>>(
      Ob, Wp, out, M_DIM, C_DIM, C_DIM);
}
